// CausalSelfAttention_59133109731929
// MI455X (gfx1250) — matrix operand order repaired and checked by reading, not compiled, not run on hardware
//
#include <hip/hip_runtime.h>
#include <hip/hip_bf16.h>

#define D_MODEL 1024
#define N_HEAD  16
#define SEQ     2048
#define BATCH   4
#define HEAD_DIM 64
#define M_TOK   (BATCH * SEQ)

typedef __attribute__((ext_vector_type(16))) __bf16 v16bf;
typedef __attribute__((ext_vector_type(8)))  float  v8f;
typedef __attribute__((ext_vector_type(4)))  unsigned int v4u;
typedef __attribute__((ext_vector_type(8)))  unsigned int v8u;

// ---------- helpers ----------

__device__ inline unsigned short f2bf(float f) {
  unsigned u = __builtin_bit_cast(unsigned, f);
  u += 0x7FFFu + ((u >> 16) & 1u);   // round-to-nearest-even
  return (unsigned short)(u >> 16);
}

__device__ inline unsigned pack2bf(float a, float b) {
  return (unsigned)f2bf(a) | ((unsigned)f2bf(b) << 16);
}

__device__ inline v8f wmma_bf16(v16bf a, v16bf b, v8f c) {
  // D = A(16x32 bf16) * B(32x16 bf16) + C(16x16 f32)
  return __builtin_amdgcn_wmma_f32_16x16x32_bf16(
      /*neg_a=*/false, a, /*neg_b=*/false, b,
      /*c_mod=*/(short)0, c, /*reuse_a=*/false, /*reuse_b=*/false);
}

// A-operand fragment: 16x32 bf16, row-major source (rows = M of A).
// Lane(lo,hi): v0..7 hold k = hi*16 + 0..15
__device__ inline v16bf load_frag_a(const unsigned short* base, int row, int ld, int k0) {
  int lane = threadIdx.x & 31;
  int lo = lane & 15, hi = lane >> 4;
  const unsigned short* p = base + (size_t)(row + lo) * ld + (k0 + hi * 16);
  v4u x = *(const v4u*)p;
  v4u y = *(const v4u*)(p + 8);
  v8u r;
  r[0]=x[0]; r[1]=x[1]; r[2]=x[2]; r[3]=x[3];
  r[4]=y[0]; r[5]=y[1]; r[6]=y[2]; r[7]=y[3];
  return __builtin_bit_cast(v16bf, r);
}

// B-operand fragment: 32x16 bf16, loaded from a row-major matrix whose rows
// are B's columns (i.e. B[k][n] = src[n][k]).
// Lane(lo=n,hi): v0..7 hold k = hi*16 + 0..15
__device__ inline v16bf load_frag_b(const unsigned short* base, int row, int ld, int k0) {
  int lane = threadIdx.x & 31;
  int lo = lane & 15, hi = lane >> 4;
  const unsigned short* p = base + (size_t)(row + lo) * ld + (k0 + hi * 16);
  v4u x = *(const v4u*)p;
  v4u y = *(const v4u*)(p + 8);
  v8u r;
  r[0]=x[0]; r[1]=x[1]; r[2]=x[2]; r[3]=x[3];
  r[4]=y[0]; r[5]=y[1]; r[6]=y[2]; r[7]=y[3];
  return __builtin_bit_cast(v16bf, r);
}

// ---------- kernels ----------

__global__ __launch_bounds__(256) void cvt_bf16_kernel(const float* __restrict__ in,
                                                       unsigned short* __restrict__ out,
                                                       int n) {
  int i = blockIdx.x * 256 + threadIdx.x;
  if (i < n) out[i] = f2bf(in[i]);
}

// C[M,N] = A[M,K](bf16) * W[N,K]^T(bf16) + bias[N]
// Block tile 128x128 (8 waves), wave tile 32x64 -> 8 WMMA per 12 b128 loads.
// mode 0: out bf16 [B,H,S,hd]      (Q, K)
// mode 1: out bf16 [B,H,hd,S]      (V transposed)
// mode 2: out f32  [M,N]           (final projection)
__global__ __launch_bounds__(256) void gemm_wmma_kernel(
    const unsigned short* __restrict__ A, const unsigned short* __restrict__ W,
    const float* __restrict__ bias, void* __restrict__ out,
    int M, int N, int K, int mode) {
  int wave = threadIdx.x >> 5;
  int lane = threadIdx.x & 31;
  int lo = lane & 15, hi = lane >> 4;
  int wm = wave & 3, wn = wave >> 2;
  int m0 = blockIdx.x * 128 + wm * 32;
  int n0 = blockIdx.y * 128 + wn * 64;

  v8f acc[2][4] = {};
  for (int k0 = 0; k0 < K; k0 += 32) {
    v16bf a0 = load_frag_a(A, m0,      K, k0);
    v16bf a1 = load_frag_a(A, m0 + 16, K, k0);
    v16bf b0 = load_frag_b(W, n0,      K, k0);
    v16bf b1 = load_frag_b(W, n0 + 16, K, k0);
    v16bf b2 = load_frag_b(W, n0 + 32, K, k0);
    v16bf b3 = load_frag_b(W, n0 + 48, K, k0);
    acc[0][0] = wmma_bf16(a0, b0, acc[0][0]);
    acc[0][1] = wmma_bf16(a0, b1, acc[0][1]);
    acc[0][2] = wmma_bf16(a0, b2, acc[0][2]);
    acc[0][3] = wmma_bf16(a0, b3, acc[0][3]);
    acc[1][0] = wmma_bf16(a1, b0, acc[1][0]);
    acc[1][1] = wmma_bf16(a1, b1, acc[1][1]);
    acc[1][2] = wmma_bf16(a1, b2, acc[1][2]);
    acc[1][3] = wmma_bf16(a1, b3, acc[1][3]);
  }

  for (int im = 0; im < 2; ++im) {
    for (int in_ = 0; in_ < 4; ++in_) {
      int n = n0 + in_ * 16 + lo;
      float bval = bias[n];
      for (int r = 0; r < 8; ++r) {
        int m = m0 + im * 16 + r + hi * 8;
        float val = acc[im][in_][r] + bval;
        if (mode == 2) {
          ((float*)out)[(size_t)m * N + n] = val;
        } else {
          int b = m >> 11, s = m & (SEQ - 1);     // m = b*SEQ + s
          int h = n >> 6,  e = n & (HEAD_DIM - 1);
          size_t idx;
          if (mode == 0)
            idx = (((size_t)(b * N_HEAD + h)) * SEQ + s) * HEAD_DIM + e;
          else
            idx = (((size_t)(b * N_HEAD + h)) * HEAD_DIM + e) * SEQ + s;
          ((unsigned short*)out)[idx] = f2bf(val);
        }
      }
    }
  }
}

// Flash attention. Q,K: [B,H,S,hd] bf16; Vt: [B,H,hd,S] bf16; Y: [B,S,D] bf16.
// grid.x = S/128, grid.y = B*H; 8 waves/block; each wave: 16 queries.
__global__ __launch_bounds__(256) void attn_wmma_kernel(
    const unsigned short* __restrict__ Q, const unsigned short* __restrict__ K,
    const unsigned short* __restrict__ Vt, unsigned short* __restrict__ Y) {
  int wave = threadIdx.x >> 5;
  int lane = threadIdx.x & 31;
  int lo = lane & 15, hi = lane >> 4;
  int bh = blockIdx.y;
  int b = bh >> 4, h = bh & (N_HEAD - 1);
  int q0 = blockIdx.x * 128 + wave * 16;

  const unsigned short* Qh = Q  + (size_t)bh * SEQ * HEAD_DIM;
  const unsigned short* Kh = K  + (size_t)bh * SEQ * HEAD_DIM;
  const unsigned short* Vh = Vt + (size_t)bh * HEAD_DIM * SEQ;

  // Q fragments, B-operand role for S^T = K * Q^T (two K-steps over hd=64)
  v16bf qf0 = load_frag_b(Qh, q0, HEAD_DIM, 0);
  v16bf qf1 = load_frag_b(Qh, q0, HEAD_DIM, 32);

  v8f o[4] = {};                 // O^T: 64(hd) x 16(q), 4 tiles of 16x16
  float mrun = -1e30f, lrun = 0.0f;
  const float sc = 0.125f * 1.44269504089f;  // 1/sqrt(64) * log2(e)
  int q = q0 + lo;               // this lane's query (as softmax column)
  int nj = (q0 + 15) / 32 + 1;   // 32-key chunks needed (causal)

  for (int j = 0; j < nj; ++j) {
    int kbase = j * 32;
    // S^T tiles: s0 = keys kbase..+15, s1 = keys kbase+16..+31 (x queries)
    v8f s0 = {}, s1 = {};
    {
      v16bf kf;
      kf = load_frag_a(Kh, kbase,      HEAD_DIM, 0);  s0 = wmma_bf16(kf, qf0, s0);
      kf = load_frag_a(Kh, kbase,      HEAD_DIM, 32); s0 = wmma_bf16(kf, qf1, s0);
      kf = load_frag_a(Kh, kbase + 16, HEAD_DIM, 0);  s1 = wmma_bf16(kf, qf0, s1);
      kf = load_frag_a(Kh, kbase + 16, HEAD_DIM, 32); s1 = wmma_bf16(kf, qf1, s1);
    }
    // scale (log2 domain) + causal mask; key index = kbase + (16t) + r + 8*hi
    bool need_mask = (kbase + 31) > q0;
    #pragma unroll
    for (int r = 0; r < 8; ++r) {
      int k0e = kbase + r + hi * 8;
      s0[r] = (need_mask && (k0e > q))       ? -3.0e38f : s0[r] * sc;
      s1[r] = (need_mask && (k0e + 16 > q))  ? -3.0e38f : s1[r] * sc;
    }
    // online softmax: per-lane max over 16, merge lane halves with one swap
    float mt = s0[0];
    #pragma unroll
    for (int r = 0; r < 8; ++r) { mt = fmaxf(mt, s0[r]); mt = fmaxf(mt, s1[r]); }
    mt = fmaxf(mt, __shfl_xor(mt, 16));
    float mnew = fmaxf(mrun, mt);
    float corr = exp2f(mrun - mnew);
    float psum = 0.0f;
    #pragma unroll
    for (int r = 0; r < 8; ++r) {
      s0[r] = exp2f(s0[r] - mnew); psum += s0[r];
      s1[r] = exp2f(s1[r] - mnew); psum += s1[r];
    }
    lrun = lrun * corr + psum;
    mrun = mnew;
    #pragma unroll
    for (int ef = 0; ef < 4; ++ef)
      #pragma unroll
      for (int r = 0; r < 8; ++r) o[ef][r] *= corr;

    // Build P B-fragment (32 keys x 16 queries) from S^T accumulators.
    // hi=0 lane owns keys 0..7 (s0) / 16..23 (s1); partner lane (^16) has 8..15 / 24..31.
    unsigned p0[4], p1[4], x0[4], x1[4];
    #pragma unroll
    for (int v = 0; v < 4; ++v) {
      p0[v] = pack2bf(s0[2 * v], s0[2 * v + 1]);
      p1[v] = pack2bf(s1[2 * v], s1[2 * v + 1]);
    }
    #pragma unroll
    for (int v = 0; v < 4; ++v) {
      x0[v] = __shfl_xor(p0[v], 16);
      x1[v] = __shfl_xor(p1[v], 16);
    }
    v8u pf;
    #pragma unroll
    for (int v = 0; v < 4; ++v) {
      pf[v]     = hi ? x1[v] : p0[v];   // k = hi*16 + 2v,2v+1
      pf[4 + v] = hi ? p1[v] : x0[v];   // k = hi*16 + 8 + 2v,2v+1
    }
    v16bf pfrag = __builtin_bit_cast(v16bf, pf);

    // O^T += V^T(16hd x 32k) * P(32k x 16q), per 16-row hd tile
    #pragma unroll
    for (int ef = 0; ef < 4; ++ef) {
      v16bf vf = load_frag_a(Vh, ef * 16, SEQ, kbase);
      o[ef] = wmma_bf16(vf, pfrag, o[ef]);
    }
  }

  // finalize: merge row-sum halves, normalize, store contiguous bf16 rows
  lrun += __shfl_xor(lrun, 16);
  float inv = 1.0f / lrun;
  size_t rowbase = ((size_t)(b * SEQ + q)) * D_MODEL + h * HEAD_DIM;
  #pragma unroll
  for (int ef = 0; ef < 4; ++ef) {
    v4u outp;
    #pragma unroll
    for (int v = 0; v < 4; ++v)
      outp[v] = pack2bf(o[ef][2 * v] * inv, o[ef][2 * v + 1] * inv);
    // e = ef*16 + hi*8 + (0..7) contiguous -> one 16B store
    *(v4u*)(Y + rowbase + ef * 16 + hi * 8) = outp;
  }
}

// ---------- launch ----------

extern "C" void kernel_launch(void* const* d_in, const int* in_sizes, int n_in,
                              void* d_out, int out_size, void* d_ws, size_t ws_size,
                              hipStream_t stream) {
  const float* x  = (const float*)d_in[0];
  const float* Wq = (const float*)d_in[1];
  const float* bq = (const float*)d_in[2];
  const float* Wk = (const float*)d_in[3];
  const float* bk = (const float*)d_in[4];
  const float* Wv = (const float*)d_in[5];
  const float* bv = (const float*)d_in[6];
  const float* Wo = (const float*)d_in[7];
  const float* bo = (const float*)d_in[8];

  char* ws = (char*)d_ws;
  size_t off = 0;
  auto take = [&](size_t bytes) {
    char* p = ws + off;
    off += (bytes + 255) & ~(size_t)255;
    return p;
  };
  unsigned short* xb  = (unsigned short*)take((size_t)M_TOK * D_MODEL * 2);
  unsigned short* wqb = (unsigned short*)take((size_t)D_MODEL * D_MODEL * 2);
  unsigned short* wkb = (unsigned short*)take((size_t)D_MODEL * D_MODEL * 2);
  unsigned short* wvb = (unsigned short*)take((size_t)D_MODEL * D_MODEL * 2);
  unsigned short* wob = (unsigned short*)take((size_t)D_MODEL * D_MODEL * 2);
  unsigned short* Qb  = (unsigned short*)take((size_t)M_TOK * D_MODEL * 2);
  unsigned short* Kb  = (unsigned short*)take((size_t)M_TOK * D_MODEL * 2);
  unsigned short* Vtb = (unsigned short*)take((size_t)M_TOK * D_MODEL * 2);
  unsigned short* Yb  = (unsigned short*)take((size_t)M_TOK * D_MODEL * 2);

  int nX = M_TOK * D_MODEL;
  int nW = D_MODEL * D_MODEL;
  cvt_bf16_kernel<<<(nX + 255) / 256, 256, 0, stream>>>(x,  xb,  nX);
  cvt_bf16_kernel<<<(nW + 255) / 256, 256, 0, stream>>>(Wq, wqb, nW);
  cvt_bf16_kernel<<<(nW + 255) / 256, 256, 0, stream>>>(Wk, wkb, nW);
  cvt_bf16_kernel<<<(nW + 255) / 256, 256, 0, stream>>>(Wv, wvb, nW);
  cvt_bf16_kernel<<<(nW + 255) / 256, 256, 0, stream>>>(Wo, wob, nW);

  dim3 gg(M_TOK / 128, D_MODEL / 128);
  gemm_wmma_kernel<<<gg, 256, 0, stream>>>(xb, wqb, bq, Qb,  M_TOK, D_MODEL, D_MODEL, 0);
  gemm_wmma_kernel<<<gg, 256, 0, stream>>>(xb, wkb, bk, Kb,  M_TOK, D_MODEL, D_MODEL, 0);
  gemm_wmma_kernel<<<gg, 256, 0, stream>>>(xb, wvb, bv, Vtb, M_TOK, D_MODEL, D_MODEL, 1);

  dim3 ga(SEQ / 128, BATCH * N_HEAD);
  attn_wmma_kernel<<<ga, 256, 0, stream>>>(Qb, Kb, Vtb, Yb);

  gemm_wmma_kernel<<<gg, 256, 0, stream>>>(Yb, wob, bo, d_out, M_TOK, D_MODEL, D_MODEL, 2);
}
